// SparseLayer_71957882077719
// MI455X (gfx1250) — matrix operand order repaired and checked by reading, not compiled, not run on hardware
//
#include <hip/hip_runtime.h>
#include <stdint.h>

typedef __attribute__((ext_vector_type(16))) _Float16 v16h;
typedef __attribute__((ext_vector_type(8)))  float    v8f;

#define OUT_DIM 1024
#define IN_DIM  1024
#define NCHUNK  128     // 4096 / 32
#define CHUNK_K 32
#define NPTS    256     // 32 gaussians * 8 points
#define BATCH   8

struct U2 { uint32_t a, b; };

__device__ __forceinline__ uint32_t rotl32(uint32_t v, int n) {
  return (v << n) | (v >> (32 - n));
}

// Threefry-2x32, 20 rounds (JAX PRNG core).
__device__ __forceinline__ U2 threefry2x32(uint32_t k0, uint32_t k1,
                                           uint32_t x0, uint32_t x1) {
  uint32_t ks2 = 0x1BD11BDAu ^ k0 ^ k1;
  x0 += k0; x1 += k1;
  x0 += x1; x1 = rotl32(x1, 13); x1 ^= x0;
  x0 += x1; x1 = rotl32(x1, 15); x1 ^= x0;
  x0 += x1; x1 = rotl32(x1, 26); x1 ^= x0;
  x0 += x1; x1 = rotl32(x1,  6); x1 ^= x0;
  x0 += k1;  x1 += ks2 + 1u;
  x0 += x1; x1 = rotl32(x1, 17); x1 ^= x0;
  x0 += x1; x1 = rotl32(x1, 29); x1 ^= x0;
  x0 += x1; x1 = rotl32(x1, 16); x1 ^= x0;
  x0 += x1; x1 = rotl32(x1, 24); x1 ^= x0;
  x0 += ks2; x1 += k0 + 2u;
  x0 += x1; x1 = rotl32(x1, 13); x1 ^= x0;
  x0 += x1; x1 = rotl32(x1, 15); x1 ^= x0;
  x0 += x1; x1 = rotl32(x1, 26); x1 ^= x0;
  x0 += x1; x1 = rotl32(x1,  6); x1 ^= x0;
  x0 += k0;  x1 += k1 + 3u;
  x0 += x1; x1 = rotl32(x1, 17); x1 ^= x0;
  x0 += x1; x1 = rotl32(x1, 29); x1 ^= x0;
  x0 += x1; x1 = rotl32(x1, 16); x1 ^= x0;
  x0 += x1; x1 = rotl32(x1, 24); x1 ^= x0;
  x0 += k1;  x1 += ks2 + 4u;
  x0 += x1; x1 = rotl32(x1, 13); x1 ^= x0;
  x0 += x1; x1 = rotl32(x1, 15); x1 ^= x0;
  x0 += x1; x1 = rotl32(x1, 26); x1 ^= x0;
  x0 += x1; x1 = rotl32(x1,  6); x1 ^= x0;
  x0 += ks2; x1 += k0 + 5u;
  U2 r; r.a = x0; r.b = x1; return r;
}

__global__ void init_y_kernel(const float* __restrict__ bias, float* __restrict__ y) {
  int i = blockIdx.x * blockDim.x + threadIdx.x;
  if (i < BATCH * OUT_DIM) y[i] = bias[i & (OUT_DIM - 1)];
}

// One workgroup (256 threads = 8 waves) per (batch, chunk).
__global__ void __launch_bounds__(256)
sparse_chunk_kernel(const float* __restrict__ x,
                    const float* __restrict__ means,
                    const float* __restrict__ sigmas,
                    const float* __restrict__ values,
                    float* __restrict__ y) {
  __shared__ __align__(16) _Float16 P16[NPTS][CHUNK_K];  // 16 KB: densities (f16 for WMMA A)
  __shared__ float    meanS[CHUNK_K][2];
  __shared__ float    inv2S[CHUNK_K][2];
  __shared__ float    valS[CHUNK_K];
  __shared__ float    colsum[CHUNK_K];
  __shared__ _Float16 w16[CHUNK_K];
  __shared__ int      codes[NPTS];
  __shared__ float    valsP[NPTS];

  const int t  = threadIdx.x;           // 0..255 : one point per thread
  const int bc = blockIdx.x;            // 0..1023
  const int b  = bc / NCHUNK;
  const int ch = bc - b * NCHUNK;
  const int gbase = b * (NCHUNK * CHUNK_K) + ch * CHUNK_K;

  // ---- stage gaussian parameters in LDS ----
  if (t < CHUNK_K * 2) {
    int k = t >> 1, r = t & 1;
    meanS[k][r] = means[(gbase + k) * 2 + r];
    inv2S[k][r] = 1.0f / (1e-6f + sigmas[(gbase + k) * 2 + r]);
  }
  if (t < CHUNK_K) { valS[t] = values[gbase + t]; colsum[t] = 0.0f; }
  __syncthreads();

  // ---- generate this thread's integer tuple ----
  const int k = t >> 3;      // gaussian within chunk
  const int v = t & 7;       // variant: 0..3 neighbors, 4..5 global, 6..7 local
  const float m0 = meanS[k][0];
  const float m1 = meanS[k][1];
  float p0, p1;
  if (v < 4) {
    // FLOOR_MASK order: (T,T),(T,F),(F,T),(F,F)
    p0 = (v < 2)        ? floorf(m0) : ceilf(m0);
    p1 = ((v & 1) == 0) ? floorf(m1) : ceilf(m1);
  } else {
    const int draw = v & 1;
    const uint32_t ctr = (uint32_t)(((bc * CHUNK_K) + k) * 2 + draw);
    U2 rnd = (v < 6) ? threefry2x32(0x9E3779B9u, 0x0000002Au, ctr, 0u)   // global stream
                     : threefry2x32(0xBB67AE85u, 0x0000002Au, ctr, 1u);  // local stream
    const float u0 = (float)(rnd.a >> 8) * (1.0f / 16777216.0f) * (1.0f - 1e-6f);
    const float u1 = (float)(rnd.b >> 8) * (1.0f / 16777216.0f) * (1.0f - 1e-6f);
    if (v < 6) {
      p0 = floorf(u0 * (float)OUT_DIM);
      p1 = floorf(u1 * (float)IN_DIM);
    } else {
      const float mns0 = rintf(m0), mns1 = rintf(m1);
      float lo0 = mns0 - 8.0f, lo1 = mns1 - 8.0f;
      const float up0 = mns0 + 8.0f, up1 = mns1 + 8.0f;
      if (lo0 < 0.0f) lo0 = 0.0f;
      if (up0 > (float)OUT_DIM) lo0 = (float)OUT_DIM - 16.0f;
      if (lo1 < 0.0f) lo1 = 0.0f;
      if (up1 > (float)IN_DIM) lo1 = (float)IN_DIM - 16.0f;
      p0 = u0 * 16.0f + lo0;   // int32 cast below truncates, matching astype(int32)
      p1 = u1 * 16.0f + lo1;
    }
  }
  int oi = (int)p0;
  int ji = (int)p1;
  oi = min(max(oi, 0), OUT_DIM - 1);
  ji = min(max(ji, 0), IN_DIM - 1);

  const int code = oi * IN_DIM + ji;
  codes[t] = code;
  __syncthreads();

  // ---- duplicate detection (earlier index with same code wins) ----
  bool dup = false;
  for (int q = 0; q < t; ++q) dup |= (codes[q] == code);

  // ---- densities: P[t][g] = exp(-0.5 * mahalanobis^2), dup rows zeroed ----
  const float fp0 = (float)oi, fp1 = (float)ji;
  #pragma unroll 4
  for (int g = 0; g < CHUNK_K; ++g) {
    const float d0 = fp0 - meanS[g][0];
    const float d1 = fp1 - meanS[g][1];
    const float d2 = d0 * d0 * inv2S[g][0] + d1 * d1 * inv2S[g][1];
    const float pr = dup ? 0.0f : __expf(-0.5f * d2);
    P16[t][g] = (_Float16)pr;
    atomicAdd(&colsum[g], pr);          // ds_add_f32
  }
  __syncthreads();

  // ---- fold per-column normalization into the B operand: w = values / colsum ----
  if (t < CHUNK_K) w16[t] = (_Float16)(valS[t] / colsum[t]);
  __syncthreads();

  // ---- vals = P (256x32) @ w via V_WMMA_F32_16X16X32_F16 (w replicated to 16 cols) ----
  const int lane = t & 31;
  const int wv   = t >> 5;

  // B 32x16 f16: lanes 0-15 hold K {0..7, 16..23} (N=lane), lanes 16-31 hold K {8..15, 24..31}.
  v16h bmat;
  {
    const int kb = (lane < 16) ? 0 : 8;
    #pragma unroll
    for (int h = 0; h < 8; ++h) { bmat[h] = w16[kb + h]; bmat[h + 8] = w16[kb + 16 + h]; }
  }

  #pragma unroll
  for (int tt = 0; tt < 2; ++tt) {
    const int tile = wv * 2 + tt;                 // 16 tiles over 8 waves
    const int row  = tile * 16 + (lane & 15);
    // A 16x32 f16 layout: lane<16 -> K {0..7, 16..23}; lane>=16 -> K {8..15, 24..31}
    const int kb = (lane < 16) ? 0 : 8;
    v16h amat;
    #pragma unroll
    for (int h = 0; h < 8; ++h) {
      amat[h]     = P16[row][kb + h];
      amat[h + 8] = P16[row][kb + 16 + h];
    }
    v8f acc = {};
    acc = __builtin_amdgcn_wmma_f32_16x16x32_f16(
        /*neg_a=*/false, amat, /*neg_b=*/false, bmat,
        /*c_mod=*/(short)0, acc, /*reuse_a=*/false, /*reuse_b=*/false);
    // D layout: lane n holds column n; VGPR q holds M=q (lanes 0-15) / M=8+q (lanes 16-31).
    // Columns are identical (w replicated): lanes 0 and 16 publish rows 0-7 / 8-15.
    if ((lane & 15) == 0) {
      const int mb = (lane >> 4) * 8;
      #pragma unroll
      for (int q = 0; q < 8; ++q) valsP[tile * 16 + mb + q] = acc[q];
    }
  }
  __syncthreads();

  // ---- scatter: y[b, oi] += vals[i] * x[b, ji]  (dup rows contribute exactly 0) ----
  const float contrib = valsP[t] * x[b * IN_DIM + ji];
  atomicAdd(&y[b * OUT_DIM + oi], contrib);       // global_atomic_add_f32
}

extern "C" void kernel_launch(void* const* d_in, const int* in_sizes, int n_in,
                              void* d_out, int out_size, void* d_ws, size_t ws_size,
                              hipStream_t stream) {
  (void)in_sizes; (void)n_in; (void)out_size; (void)d_ws; (void)ws_size;
  const float* x      = (const float*)d_in[0];
  const float* means  = (const float*)d_in[1];
  const float* sigmas = (const float*)d_in[2];
  const float* values = (const float*)d_in[3];
  const float* bias   = (const float*)d_in[4];
  float* y = (float*)d_out;

  init_y_kernel<<<(BATCH * OUT_DIM + 255) / 256, 256, 0, stream>>>(bias, y);
  sparse_chunk_kernel<<<BATCH * NCHUNK, 256, 0, stream>>>(x, means, sigmas, values, y);
}
